// MultigrainTargetAttention_79173427134505
// MI455X (gfx1250) — matrix operand order repaired and checked by reading, not compiled, not run on hardware
//
#include <hip/hip_runtime.h>
#include <hip/hip_bf16.h>

typedef __attribute__((ext_vector_type(16))) _Float16 v16h;
typedef __attribute__((ext_vector_type(8)))  _Float16 v8h;
typedef __attribute__((ext_vector_type(8)))  float    v8f;

#define DIMX 128
#define NB   64
#define NS   50
#define NL   20

__device__ __forceinline__ float red_full32(float v) {
#pragma unroll
  for (int m = 1; m < 32; m <<= 1) v += __shfl_xor(v, m, 32);
  return v;
}
__device__ __forceinline__ float red_half16(float v) {
#pragma unroll
  for (int m = 1; m < 16; m <<= 1) v += __shfl_xor(v, m, 32);
  return v;
}
__device__ __forceinline__ float redmax32(float v) {
#pragma unroll
  for (int m = 1; m < 32; m <<= 1) v = fmaxf(v, __shfl_xor(v, m, 32));
  return v;
}

// Convert the 3 surviving (KEEP==1, singleton-subset) weight matrices to f16.
__global__ void __launch_bounds__(256) mgta_prep(const float* __restrict__ glu_w,
                                                 const float* __restrict__ glu2_w,
                                                 _Float16* __restrict__ W1h,
                                                 _Float16* __restrict__ W2h) {
  const int per = DIMX * DIMX;
  int i = blockIdx.x * 256 + threadIdx.x;
  if (i >= 3 * per) return;
  int g = i / per;
  int rem = i - g * per;
  int k = (g == 2) ? 3 : g;   // singleton subsets map to K-index {0,1,3}
  W1h[i] = (_Float16)glu_w[(size_t)k * per + rem];
  W2h[i] = (_Float16)glu2_w[(size_t)k * per + rem];
}

__global__ void __launch_bounds__(256) mgta_main(
    const float* __restrict__ sess, const float* __restrict__ target,
    const float* __restrict__ maskp, const float* __restrict__ pos,
    const float* __restrict__ q_w, const float* __restrict__ glu_b,
    const _Float16* __restrict__ W1h, const _Float16* __restrict__ W2h,
    float* __restrict__ out) {

  __shared__ __align__(32) _Float16 sX1[32 * DIMX];   // item + pos  (f16, rows 20..31 zero)
  __shared__ __align__(32) _Float16 sX2[32 * DIMX];   // item * tar  (f16)
  __shared__ __align__(16) float    sItem[NL * DIMX]; // normalized item (f32)
  __shared__ __align__(16) float    sTarn[DIMX];
  __shared__ __align__(16) float    sTar0[DIMX];
  __shared__ float sLogit[32];
  __shared__ float sBeta[32];
  __shared__ float sSess[DIMX];

  const int tid  = threadIdx.x;
  const int wave = tid >> 5;
  const int lane = tid & 31;
  const int r    = lane & 15;
  const int hf   = lane >> 4;

  const int bs = blockIdx.x;
  const int b  = bs / NS;
  const int s  = bs - b * NS;

  // one-time: zero-pad A rows 20..31, zero sess accumulator
  for (int i = tid; i < 12 * DIMX; i += 256) {
    sX1[NL * DIMX + i] = (_Float16)0.0f;
    sX2[NL * DIMX + i] = (_Float16)0.0f;
  }
  if (tid < DIMX) sSess[tid] = 0.0f;

  for (int g = 0; g < 3; ++g) {
    const int k6 = (g == 2) ? 3 : g;

    // ---- normalize target (wave0); zero logits (wave1)
    if (tid < 32) {
      const float* tp = target + (((size_t)g * NB + b) * NS + s) * DIMX;
      float4 t4 = *(const float4*)(tp + 4 * lane);
      float ss = t4.x*t4.x + t4.y*t4.y + t4.z*t4.z + t4.w*t4.w;
      ss = red_full32(ss);
      float inv = 1.0f / fmaxf(sqrtf(ss), 1e-12f);
      float tv[4] = {t4.x*inv, t4.y*inv, t4.z*inv, t4.w*inv};
#pragma unroll
      for (int u = 0; u < 4; ++u) {
        sTarn[4*lane+u] = tv[u];
        if (g == 0) sTar0[4*lane+u] = tv[u];
      }
    } else if (tid < 64) {
      sLogit[tid - 32] = 0.0f;
    }
    __syncthreads();

    // ---- normalize item rows, build f16 GEMM operands in LDS
    const float* srow = sess + ((size_t)g * NB + b) * (size_t)(NS * NL * DIMX)
                             + (size_t)s * (NL * DIMX);
    for (int j = wave; j < NL; j += 8) {
      const float4 x4 = *(const float4*)(srow + j * DIMX + 4 * lane);
      float ss = x4.x*x4.x + x4.y*x4.y + x4.z*x4.z + x4.w*x4.w;
      ss = red_full32(ss);
      float inv = 1.0f / fmaxf(sqrtf(ss), 1e-12f);
      const float4 p4 = *(const float4*)(pos + j * DIMX + 4 * lane);
      float xv[4] = {x4.x, x4.y, x4.z, x4.w};
      float pv[4] = {p4.x, p4.y, p4.z, p4.w};
#pragma unroll
      for (int u = 0; u < 4; ++u) {
        int d = 4 * lane + u;
        float xi = xv[u] * inv;
        sItem[j * DIMX + d] = xi;
        sX1[j * DIMX + d] = (_Float16)(xi + pv[u]);
        sX2[j * DIMX + d] = (_Float16)(xi * sTarn[d]);
      }
    }
    __syncthreads();

    // ---- WMMA GEMM: pre = X1*W1^T + X2*W2^T ; wave handles N-tile n0, M-tiles 0 and 1
    const _Float16* W1g = W1h + (size_t)g * DIMX * DIMX;
    const _Float16* W2g = W2h + (size_t)g * DIMX * DIMX;
    const int n0 = wave * 16;
    v8f acc0 = {};
    v8f acc1 = {};
#pragma unroll
    for (int kk = 0; kk < 4; ++kk) {
      const int kb = kk * 32;
      const int ao = kb + hf * 8;    // A-frag: K {ao..ao+7} U {ao+16..ao+23}
      const int bo = kb + hf * 8;    // B-frag: K {bo..bo+7} U {bo+16..bo+23}
      v8h a0lo = *(const v8h*)&sX1[r * DIMX + ao];
      v8h a0hi = *(const v8h*)&sX1[r * DIMX + ao + 16];
      v8h a1lo = *(const v8h*)&sX1[(16 + r) * DIMX + ao];
      v8h a1hi = *(const v8h*)&sX1[(16 + r) * DIMX + ao + 16];
      v16h a0 = __builtin_shufflevector(a0lo, a0hi, 0,1,2,3,4,5,6,7,8,9,10,11,12,13,14,15);
      v16h a1 = __builtin_shufflevector(a1lo, a1hi, 0,1,2,3,4,5,6,7,8,9,10,11,12,13,14,15);
      v16h bf1 = __builtin_shufflevector(*(const v8h*)&W1g[(n0 + r) * DIMX + bo], *(const v8h*)&W1g[(n0 + r) * DIMX + bo + 16], 0,1,2,3,4,5,6,7,8,9,10,11,12,13,14,15);
      acc0 = __builtin_amdgcn_wmma_f32_16x16x32_f16(false, a0, false, bf1, (short)0, acc0, false, false);
      acc1 = __builtin_amdgcn_wmma_f32_16x16x32_f16(false, a1, false, bf1, (short)0, acc1, false, false);

      v8h c0lo = *(const v8h*)&sX2[r * DIMX + ao];
      v8h c0hi = *(const v8h*)&sX2[r * DIMX + ao + 16];
      v8h c1lo = *(const v8h*)&sX2[(16 + r) * DIMX + ao];
      v8h c1hi = *(const v8h*)&sX2[(16 + r) * DIMX + ao + 16];
      v16h c0 = __builtin_shufflevector(c0lo, c0hi, 0,1,2,3,4,5,6,7,8,9,10,11,12,13,14,15);
      v16h c1 = __builtin_shufflevector(c1lo, c1hi, 0,1,2,3,4,5,6,7,8,9,10,11,12,13,14,15);
      v16h bf2 = __builtin_shufflevector(*(const v8h*)&W2g[(n0 + r) * DIMX + bo], *(const v8h*)&W2g[(n0 + r) * DIMX + bo + 16], 0,1,2,3,4,5,6,7,8,9,10,11,12,13,14,15);
      acc0 = __builtin_amdgcn_wmma_f32_16x16x32_f16(false, c0, false, bf2, (short)0, acc0, false, false);
      acc1 = __builtin_amdgcn_wmma_f32_16x16x32_f16(false, c1, false, bf2, (short)0, acc1, false, false);
    }

    // ---- epilogue: sigmoid(pre + bias) * q[e], reduce over e into logit[j]
    {
      const int e = n0 + r;
      const float bias = glu_b[(size_t)k6 * DIMX + e];
      const float qv   = q_w[(size_t)k6 * DIMX + e];
#pragma unroll
      for (int v = 0; v < 8; ++v) {
        float g0 = qv / (1.0f + __expf(-(acc0[v] + bias)));
        float g1 = qv / (1.0f + __expf(-(acc1[v] + bias)));
        g0 = red_half16(g0);
        g1 = red_half16(g1);
        if (r == 0) {
          int j0 = v + 8 * hf;          // 0..15 (always valid)
          atomicAdd(&sLogit[j0], g0);
          int j1 = 16 + v + 8 * hf;     // 16..31
          if (j1 < NL) atomicAdd(&sLogit[j1], g1);
        }
      }
    }
    __syncthreads();

    // ---- softmax over L (wave0)
    if (tid < 32) {
      float mk = (tid < NL) ? maskp[(size_t)b * NL + tid] : 0.0f;
      float x  = (tid < NL) ? sLogit[tid] * mk : -3.0e38f;
      float m  = redmax32(x);
      float e  = (tid < NL) ? __expf(x - m) : 0.0f;
      float sm = red_full32(e);
      sBeta[tid] = e / sm;
    }
    __syncthreads();

    // ---- sess_total += sum_j beta[j] * item[j,:]
    if (tid < DIMX) {
      float a = 0.0f;
#pragma unroll 4
      for (int j = 0; j < NL; ++j) a += sBeta[j] * sItem[j * DIMX + tid];
      sSess[tid] += a;
    }
    __syncthreads();
  }

  // ---- score = 10 * dot(l2n(sess_total), tar0n)
  if (tid < 32) {
    float ss = 0.0f, dt = 0.0f;
#pragma unroll
    for (int u = 0; u < 4; ++u) {
      float v = sSess[4 * lane + u];
      ss += v * v;
      dt += v * sTar0[4 * lane + u];
    }
    ss = red_full32(ss);
    dt = red_full32(dt);
    if (lane == 0) {
      float inv = 1.0f / fmaxf(sqrtf(ss), 1e-12f);
      out[bs] = 10.0f * dt * inv;
    }
  }
}

extern "C" void kernel_launch(void* const* d_in, const int* in_sizes, int n_in,
                              void* d_out, int out_size, void* d_ws, size_t ws_size,
                              hipStream_t stream) {
  (void)in_sizes; (void)n_in; (void)out_size; (void)ws_size;
  const float* sess   = (const float*)d_in[0];
  const float* target = (const float*)d_in[1];
  const float* maskp  = (const float*)d_in[2];
  const float* pos    = (const float*)d_in[3];
  const float* q_w    = (const float*)d_in[4];
  const float* glu_w  = (const float*)d_in[5];
  const float* glu_b  = (const float*)d_in[6];
  const float* glu2_w = (const float*)d_in[7];
  float* out = (float*)d_out;

  _Float16* W1h = (_Float16*)d_ws;                 // 3*128*128 f16
  _Float16* W2h = W1h + 3 * DIMX * DIMX;           // 3*128*128 f16

  mgta_prep<<<(3 * DIMX * DIMX + 255) / 256, 256, 0, stream>>>(glu_w, glu2_w, W1h, W2h);
  mgta_main<<<NB * NS, 256, 0, stream>>>(sess, target, maskp, pos, q_w, glu_b, W1h, W2h, out);
}
